// PerceiverResampler_3968549782270
// MI455X (gfx1250) — matrix operand order repaired and checked by reading, not compiled, not run on hardware
//
#include <hip/hip_runtime.h>
#include <math.h>

// ---------------- problem constants ----------------
#define B_     32
#define F_     2048
#define DIM_   1024
#define H_     16
#define DH_    64
#define Q_     64
#define DEPTH_ 6
#define INNER_ 1024
#define MULT_  2

typedef unsigned short u16;
typedef unsigned int   u32;
typedef unsigned long long u64;

typedef __bf16 v16bf __attribute__((ext_vector_type(16)));
typedef float  v8f   __attribute__((ext_vector_type(8)));
typedef unsigned int v4u __attribute__((ext_vector_type(4)));
typedef int v4i __attribute__((ext_vector_type(4)));
typedef int v8i __attribute__((ext_vector_type(8)));

#if defined(__gfx1250__) && __has_builtin(__builtin_amdgcn_tensor_load_to_lds) && __has_builtin(__builtin_amdgcn_s_wait_tensorcnt)
#define USE_TDM 1
#else
#define USE_TDM 0
#endif

union AF { v16bf v; u32 u[8]; };

__device__ __forceinline__ u16 f2bf(float f) {
    union { float f; u32 u; } x; x.f = f;
    u32 r = x.u + 0x7FFFu + ((x.u >> 16) & 1u);
    return (u16)(r >> 16);
}
__device__ __forceinline__ float bf2f(u16 h) {
    union { u32 u; float f; } x; x.u = ((u32)h) << 16;
    return x.f;
}

#define NEG_INF (-__builtin_inff())

// =====================================================================
// Row statistics: mean + rstd per row
// =====================================================================
template<bool BF>
__global__ __launch_bounds__(256)
void rowstats_kernel(const void* __restrict__ X, float* __restrict__ stats, int C)
{
    __shared__ float s1[256], s2[256];
    const int row = blockIdx.x, tid = threadIdx.x;
    float a = 0.f, q = 0.f;
    for (int c = tid; c < C; c += 256) {
        float v = BF ? bf2f(((const u16*)X)[(size_t)row * C + c])
                     : ((const float*)X)[(size_t)row * C + c];
        a += v; q += v * v;
    }
    s1[tid] = a; s2[tid] = q;
    __syncthreads();
    for (int st = 128; st > 0; st >>= 1) {
        if (tid < st) { s1[tid] += s1[tid + st]; s2[tid] += s2[tid + st]; }
        __syncthreads();
    }
    if (tid == 0) {
        float m = s1[0] / (float)C;
        float var = s2[0] / (float)C - m * m;
        stats[2 * row]     = m;
        stats[2 * row + 1] = rsqrtf(var + 1e-5f);
    }
}

// =====================================================================
// In-place bf16 row normalization: X[r][c] = (X[r][c]-mean[r])*rstd[r]
// (C fixed = INNER_ so row = i >> 10)
// =====================================================================
__global__ __launch_bounds__(256)
void norm_bf16_kernel(u16* __restrict__ X, const float* __restrict__ stats)
{
    const int i = blockIdx.x * 256 + threadIdx.x;
    const int row = i >> 10;
    const float m = stats[2 * row], r = stats[2 * row + 1];
    X[i] = f2bf((bf2f(X[i]) - m) * r);
}

// =====================================================================
// Weight prep, TRANSPOSED output:  Wt[n][d] = smul * g[d] * W[d][n]  (bf16)
// Output is N x D row-major (i.e. GEMM-B stored N-major, K contiguous).
// =====================================================================
__global__ __launch_bounds__(256)
void scale_w_kernel(const float* __restrict__ W, const float* __restrict__ g,
                    u16* __restrict__ out, int D, int N, float smul)
{
    int idx = blockIdx.x * 256 + threadIdx.x;
    if (idx < D * N) {
        int d = idx / N;
        int n = idx - d * N;
        float gv = g ? g[d] : 1.f;
        out[(size_t)n * D + d] = f2bf(W[idx] * gv * smul);
    }
}

// bias[n] = smul * sum_d b[d] * W[d][n]
__global__ __launch_bounds__(256)
void bias_gemv_kernel(const float* __restrict__ W, const float* __restrict__ bvec,
                      float* __restrict__ bias, int D, int N, float smul)
{
    int n = blockIdx.x * 256 + threadIdx.x;
    if (n < N) {
        float s = 0.f;
        for (int d = 0; d < D; d++) s += bvec[d] * W[(size_t)d * N + n];
        bias[n] = s * smul;
    }
}

// x[b][q][c] = latents[q][c]
__global__ __launch_bounds__(256)
void bcast_kernel(const float* __restrict__ lat, float* __restrict__ x, int total, int per)
{
    int i = blockIdx.x * 256 + threadIdx.x;
    if (i < total) x[i] = lat[i % per];
}

// =====================================================================
// Tiled WMMA GEMM:  Out = epilogue( normA(A) @ Wt^T + bias )
//  A : M x K  (f32 or bf16, optional per-row (x-mean)*rstd)
//  Wt: N x K  bf16 row-major (pre-transposed weights; K contiguous)
//  128x128x32 tile, 256 threads (8 waves), wave does 32x64 (2x4 WMMA tiles)
//  When A is raw bf16 (no normalization): Tensor Data Mover stages both
//  tiles into double-buffered LDS (TENSORcnt pipelined, depth 2).
// =====================================================================
template<bool ABF, bool NORMA, bool OBF, bool BIAS, bool RELU, bool RES>
__global__ __launch_bounds__(256)
void gemm_wmma_kernel(const void* __restrict__ Ap, const float* __restrict__ stats,
                      const u16* __restrict__ Wt, const float* __restrict__ bias,
                      void* __restrict__ Op, const float* __restrict__ Res,
                      int M, int N, int K)
{
    constexpr bool ASYNC = ABF && !NORMA && (USE_TDM != 0);

    // stride 48 u16 = 96B (16B aligned rows; matches TDM pad 16dw data + 8dw pad)
    __shared__ __align__(16) u16 As[2][128][48];
    __shared__ __align__(16) u16 Bs[2][128][48];

    const int tid  = threadIdx.x;
    const int wv   = tid >> 5;
    const int lane = tid & 31;
    const int lh   = lane >> 4;
    const int ln   = lane & 15;
    const int bm   = blockIdx.x * 128;
    const int bn   = blockIdx.y * 128;
    const int rw   = (wv >> 1) * 32;
    const int cw   = (wv & 1) * 64;

    const v8f vzero = {0.f, 0.f, 0.f, 0.f, 0.f, 0.f, 0.f, 0.f};
    v8f acc[2][4];
    #pragma unroll
    for (int a = 0; a < 2; a++)
        #pragma unroll
        for (int b = 0; b < 4; b++) acc[a][b] = vzero;

    auto compute_step = [&](int buf) {
        AF af[2];
        #pragma unroll
        for (int ta = 0; ta < 2; ta++) {
            const int am = rw + ta * 16 + ln;
            #pragma unroll
            for (int j = 0; j < 8; j++) {
                const int k = (j < 4) ? (lh * 8 + 2 * j) : (16 + lh * 8 + 2 * (j - 4));
                af[ta].u[j] = *(const u32*)&As[buf][am][k];
            }
        }
        #pragma unroll
        for (int tb = 0; tb < 4; tb++) {
            AF bf_;
            const int nn = cw + tb * 16 + ln;
            #pragma unroll
            for (int j = 0; j < 8; j++)
                bf_.u[j] = *(const u32*)&Bs[buf][nn][(j < 4) ? (lh * 8 + 2 * j) : (16 + lh * 8 + 2 * (j - 4))];
            #pragma unroll
            for (int ta = 0; ta < 2; ta++)
                acc[ta][tb] = __builtin_amdgcn_wmma_f32_16x16x32_bf16(
                    false, af[ta].v, false, bf_.v, (short)0, acc[ta][tb], false, false);
        }
    };

    if (ASYNC) {
#if USE_TDM
        // ---- TDM descriptor group1 (shape): tile 32(k) x 128(rows), data 2B,
        //      pad_enable, interval 16 dwords (one 64B row), amount 8 dwords ----
        const u32 g1w0 = (1u << 16) | (1u << 20) | (3u << 22) | (7u << 25);
        auto make_g1 = [&](u32 dim1) -> v8i {
            v8i g;
            g[0] = (int)g1w0;
            g[1] = (int)(((u32)K & 0xFFFFu) << 16);                       // tensor_dim0 lo
            g[2] = (int)((((u32)K >> 16) & 0xFFFFu) | ((dim1 & 0xFFFFu) << 16));
            g[3] = (int)(((dim1 >> 16) & 0xFFFFu) | (32u << 16));         // tile_dim0 = 32
            g[4] = (int)128u;                                             // tile_dim1 = 128
            g[5] = (int)(u32)K;                                           // stride0 lo
            g[6] = 0;
            g[7] = 0;
            return g;
        };
        const v8i g1A = make_g1((u32)M);
        const v8i g1B = make_g1((u32)N);

        auto tdm_issue = [&](u64 gaddr, u32 laddr, const v8i& g1) {
            v4u g0;
            g0.x = 1u;                                                    // count = 1
            g0.y = laddr;                                                 // lds_addr
            g0.z = (u32)gaddr;                                            // global_addr lo
            g0.w = (u32)((gaddr >> 32) & 0x01FFFFFFu) | (2u << 30);       // hi | type=2
            v4i z4 = {0, 0, 0, 0};
#if __clang_major__ >= 23
            v8i z8 = {0, 0, 0, 0, 0, 0, 0, 0};
            __builtin_amdgcn_tensor_load_to_lds(g0, g1, z4, z4, z8, 0);
#else
            __builtin_amdgcn_tensor_load_to_lds(g0, g1, z4, z4, 0);
#endif
        };

        const u64 abase = (u64)(uintptr_t)Ap + (((u64)bm * (u64)K) << 1);
        const u64 bbase = (u64)(uintptr_t)Wt + (((u64)bn * (u64)K) << 1);
        const u32 ldsA0 = (u32)(uintptr_t)&As[0][0][0];
        const u32 ldsB0 = (u32)(uintptr_t)&Bs[0][0][0];
        const u32 BUFB  = 128u * 48u * 2u;

        if (wv == 0) {
            tdm_issue(abase, ldsA0, g1A);
            tdm_issue(bbase, ldsB0, g1B);
        }
        const int nstep = K / 32;
        for (int s = 0; s < nstep; s++) {
            const int cur = s & 1;
            if (wv == 0) {
                if (s + 1 < nstep) {
                    const u64 koff = ((u64)(s + 1) * 32u) << 1;
                    tdm_issue(abase + koff, ldsA0 + (u32)(cur ^ 1) * BUFB, g1A);
                    tdm_issue(bbase + koff, ldsB0 + (u32)(cur ^ 1) * BUFB, g1B);
                    __builtin_amdgcn_s_wait_tensorcnt(2);
                } else {
                    __builtin_amdgcn_s_wait_tensorcnt(0);
                }
            }
            __syncthreads();
            compute_step(cur);
            __syncthreads();
        }
#endif
    } else {
        // ---- synchronous VGPR staging (A may be f32 / normalized) ----
        const int arow = tid >> 1;
        const int acol = (tid & 1) * 16;
        const int brow = tid >> 1;             // Bs row (n)
        const int bcol = (tid & 1) * 16;       // k offset

        float amean = 0.f, arstd = 1.f;
        if (NORMA) {
            amean = stats[(size_t)(bm + arow) * 2];
            arstd = stats[(size_t)(bm + arow) * 2 + 1];
        }

        for (int kb = 0; kb < K; kb += 32) {
            // A tile
            if (ABF) {
                const u16* ap = (const u16*)Ap + (size_t)(bm + arow) * K + kb + acol;
                alignas(16) u16 t[16];
                *(uint4*)&t[0] = *(const uint4*)ap;
                *(uint4*)&t[8] = *(const uint4*)(ap + 8);
                #pragma unroll
                for (int j = 0; j < 16; j++) {
                    if (NORMA) As[0][arow][acol + j] = f2bf((bf2f(t[j]) - amean) * arstd);
                    else       As[0][arow][acol + j] = t[j];
                }
            } else {
                const float* ap = (const float*)Ap + (size_t)(bm + arow) * K + kb + acol;
                #pragma unroll
                for (int q = 0; q < 4; q++) {
                    float4 r = *(const float4*)(ap + q * 4);
                    float vv[4] = {r.x, r.y, r.z, r.w};
                    #pragma unroll
                    for (int t = 0; t < 4; t++) {
                        float f = vv[t];
                        if (NORMA) f = (f - amean) * arstd;
                        As[0][arow][acol + q * 4 + t] = f2bf(f);
                    }
                }
            }
            // B tile: straight row copy (weights pre-transposed N x K)
            {
                const u16* bp = Wt + (size_t)(bn + brow) * K + kb + bcol;
                *(uint4*)&Bs[0][brow][bcol]     = *(const uint4*)bp;
                *(uint4*)&Bs[0][brow][bcol + 8] = *(const uint4*)(bp + 8);
            }
            // lookahead prefetch of next K-slice
            if (kb + 32 < K) {
                if (ABF) __builtin_prefetch((const u16*)Ap + (size_t)(bm + arow) * K + kb + 32 + acol, 0, 3);
                else     __builtin_prefetch((const float*)Ap + (size_t)(bm + arow) * K + kb + 32 + acol, 0, 3);
                __builtin_prefetch(Wt + (size_t)(bn + brow) * K + kb + 32 + bcol, 0, 3);
            }
            __syncthreads();
            compute_step(0);
            __syncthreads();
        }
    }

    // ---- epilogue ----
    #pragma unroll
    for (int tb = 0; tb < 4; tb++) {
        const int gn = bn + cw + tb * 16 + ln;
        float bv = BIAS ? bias[gn] : 0.f;
        #pragma unroll
        for (int ta = 0; ta < 2; ta++) {
            #pragma unroll
            for (int j = 0; j < 8; j++) {
                const int gm = bm + rw + ta * 16 + lh * 8 + j;
                float v = acc[ta][tb][j] + bv;
                if (RELU) v = v > 0.f ? v : 0.f;
                if (RES)  v += Res[(size_t)gm * N + gn];
                if (OBF) ((u16*)Op)[(size_t)gm * N + gn] = f2bf(v);
                else     ((float*)Op)[(size_t)gm * N + gn] = v;
            }
        }
    }
}

// =====================================================================
// Flash attention per (b,h): Q=64 queries, DH=64, stream F in tiles of 64.
// 128 threads = 4 waves; wave owns 16 query rows.
// =====================================================================
__global__ __launch_bounds__(128)
void attn_kernel(const u16* __restrict__ Qm, const u16* __restrict__ Km,
                 const u16* __restrict__ Vm, const int* __restrict__ mask,
                 float* __restrict__ Out)
{
    __shared__ __align__(16) u16 qs[64][72];
    __shared__ __align__(16) u16 ks[64][72];   // [f_local][dh]
    __shared__ __align__(16) u16 vt[64][72];   // transposed: [dh][f_local]
    __shared__ __align__(16) u16 ps[64][72];   // P tile [q][f_local]
    __shared__ int msk[64];

    const int h = blockIdx.x, b = blockIdx.y;
    const int tid  = threadIdx.x;
    const int wv   = tid >> 5;
    const int lane = tid & 31;
    const int lh   = lane >> 4;
    const int ln   = lane & 15;

    // ---- stage q (64x64) ----
    {
        const int row = tid >> 1, c0 = (tid & 1) * 32;
        const u16* src = Qm + (size_t)(b * Q_ + row) * INNER_ + h * DH_ + c0;
        uint4* dst = (uint4*)&qs[row][c0];
        const uint4* s4 = (const uint4*)src;
        dst[0] = s4[0]; dst[1] = s4[1]; dst[2] = s4[2]; dst[3] = s4[3];
    }
    __syncthreads();

    AF qf[2];
    #pragma unroll
    for (int kk = 0; kk < 2; kk++)
        #pragma unroll
        for (int j = 0; j < 8; j++) {
            const int k = kk * 32 + ((j < 4) ? (lh * 8 + 2 * j) : (16 + lh * 8 + 2 * (j - 4)));
            qf[kk].u[j] = *(const u32*)&qs[wv * 16 + ln][k];
        }

    float mrow[8], lrow[8];
    #pragma unroll
    for (int j = 0; j < 8; j++) { mrow[j] = NEG_INF; lrow[j] = 0.f; }
    const v8f vzero = {0.f, 0.f, 0.f, 0.f, 0.f, 0.f, 0.f, 0.f};
    v8f accO[4];
    #pragma unroll
    for (int t = 0; t < 4; t++) accO[t] = vzero;

    for (int ft = 0; ft < F_ / 64; ft++) {
        // ---- stage K tile, V tile (transposed), mask ----
        {
            const int row = tid >> 1, c0 = (tid & 1) * 32;
            const u16* ksrc = Km + (size_t)(b * F_ + ft * 64 + row) * INNER_ + h * DH_ + c0;
            uint4* kdst = (uint4*)&ks[row][c0];
            const uint4* k4 = (const uint4*)ksrc;
            kdst[0] = k4[0]; kdst[1] = k4[1]; kdst[2] = k4[2]; kdst[3] = k4[3];

            const u16* vsrc = Vm + (size_t)(b * F_ + ft * 64 + row) * INNER_ + h * DH_ + c0;
            alignas(16) u16 t[32];
            const uint4* v4 = (const uint4*)vsrc;
            *(uint4*)&t[0]  = v4[0]; *(uint4*)&t[8]  = v4[1];
            *(uint4*)&t[16] = v4[2]; *(uint4*)&t[24] = v4[3];
            #pragma unroll
            for (int j = 0; j < 32; j++) vt[c0 + j][row] = t[j];

            if (ft + 1 < F_ / 64) {
                __builtin_prefetch(Km + (size_t)(b * F_ + (ft + 1) * 64 + row) * INNER_ + h * DH_ + c0, 0, 3);
                __builtin_prefetch(Vm + (size_t)(b * F_ + (ft + 1) * 64 + row) * INNER_ + h * DH_ + c0, 0, 3);
            }
        }
        if (tid < 64) msk[tid] = mask[b * F_ + ft * 64 + tid];
        __syncthreads();

        // ---- S = q @ k^T ----
        v8f sacc[4];
        #pragma unroll
        for (int t = 0; t < 4; t++) sacc[t] = vzero;
        #pragma unroll
        for (int kk = 0; kk < 2; kk++) {
            #pragma unroll
            for (int tb = 0; tb < 4; tb++) {
                AF bf_;
                #pragma unroll
                for (int j = 0; j < 8; j++)
                    bf_.u[j] = *(const u32*)&ks[tb * 16 + ln][kk * 32 + ((j < 4) ? (lh * 8 + 2 * j) : (16 + lh * 8 + 2 * (j - 4)))];
                sacc[tb] = __builtin_amdgcn_wmma_f32_16x16x32_bf16(
                    false, qf[kk].v, false, bf_.v, (short)0, sacc[tb], false, false);
            }
        }

        // ---- masked online softmax over this 64-col tile ----
        float pv[4][8];
        float rmax[8];
        #pragma unroll
        for (int j = 0; j < 8; j++) rmax[j] = NEG_INF;
        #pragma unroll
        for (int tb = 0; tb < 4; tb++) {
            const int kp = msk[tb * 16 + ln];
            #pragma unroll
            for (int j = 0; j < 8; j++) {
                float s = kp ? sacc[tb][j] : NEG_INF;
                pv[tb][j] = s;
                rmax[j] = fmaxf(rmax[j], s);
            }
        }
        #pragma unroll
        for (int j = 0; j < 8; j++) {
            rmax[j] = fmaxf(rmax[j], __shfl_xor(rmax[j], 1, 32));
            rmax[j] = fmaxf(rmax[j], __shfl_xor(rmax[j], 2, 32));
            rmax[j] = fmaxf(rmax[j], __shfl_xor(rmax[j], 4, 32));
            rmax[j] = fmaxf(rmax[j], __shfl_xor(rmax[j], 8, 32));
        }
        #pragma unroll
        for (int j = 0; j < 8; j++) {
            const float mo = mrow[j];
            const float mn = fmaxf(mo, rmax[j]);
            const float corr = (mn == NEG_INF) ? 1.f : __expf(mo - mn);
            float psum = 0.f;
            #pragma unroll
            for (int tb = 0; tb < 4; tb++) {
                float p = (pv[tb][j] == NEG_INF) ? 0.f : __expf(pv[tb][j] - mn);
                pv[tb][j] = p;
                psum += p;
            }
            psum += __shfl_xor(psum, 1, 32);
            psum += __shfl_xor(psum, 2, 32);
            psum += __shfl_xor(psum, 4, 32);
            psum += __shfl_xor(psum, 8, 32);
            mrow[j] = mn;
            lrow[j] = lrow[j] * corr + psum;
            #pragma unroll
            for (int tb = 0; tb < 4; tb++) accO[tb][j] *= corr;
            const int qr = wv * 16 + lh * 8 + j;
            #pragma unroll
            for (int tb = 0; tb < 4; tb++) ps[qr][tb * 16 + ln] = f2bf(pv[tb][j]);
        }
        __syncthreads();

        // ---- O += P @ V ----
        #pragma unroll
        for (int kk = 0; kk < 2; kk++) {
            AF pa;
            const int am = wv * 16 + ln;
            #pragma unroll
            for (int j = 0; j < 8; j++) {
                const int k = kk * 32 + ((j < 4) ? (lh * 8 + 2 * j) : (16 + lh * 8 + 2 * (j - 4)));
                pa.u[j] = *(const u32*)&ps[am][k];
            }
            #pragma unroll
            for (int tb = 0; tb < 4; tb++) {
                AF vb;
                #pragma unroll
                for (int j = 0; j < 8; j++)
                    vb.u[j] = *(const u32*)&vt[tb * 16 + ln][kk * 32 + ((j < 4) ? (lh * 8 + 2 * j) : (16 + lh * 8 + 2 * (j - 4)))];
                accO[tb] = __builtin_amdgcn_wmma_f32_16x16x32_bf16(
                    false, pa.v, false, vb.v, (short)0, accO[tb], false, false);
            }
        }
        __syncthreads();
    }

    // ---- finalize ----
    #pragma unroll
    for (int tb = 0; tb < 4; tb++) {
        #pragma unroll
        for (int j = 0; j < 8; j++) {
            const int qr = wv * 16 + lh * 8 + j;
            const float l_ = lrow[j];
            const float inv = (l_ > 0.f) ? 1.f / l_ : 0.f;
            Out[(size_t)(b * Q_ + qr) * INNER_ + h * DH_ + tb * 16 + ln] = accO[tb][j] * inv;
        }
    }
}

// =====================================================================
// Final LN (writes fp32 result)
// =====================================================================
__global__ __launch_bounds__(256)
void ln_final_kernel(const float* __restrict__ X, const float* __restrict__ g,
                     const float* __restrict__ bb, float* __restrict__ out, int C)
{
    __shared__ float s1[256], s2[256];
    __shared__ float smean, srstd;
    const int row = blockIdx.x, tid = threadIdx.x;
    float a = 0.f, q = 0.f;
    for (int c = tid; c < C; c += 256) {
        float v = X[(size_t)row * C + c];
        a += v; q += v * v;
    }
    s1[tid] = a; s2[tid] = q;
    __syncthreads();
    for (int st = 128; st > 0; st >>= 1) {
        if (tid < st) { s1[tid] += s1[tid + st]; s2[tid] += s2[tid + st]; }
        __syncthreads();
    }
    if (tid == 0) {
        float m = s1[0] / (float)C;
        smean = m;
        srstd = rsqrtf(s2[0] / (float)C - m * m + 1e-5f);
    }
    __syncthreads();
    const float m = smean, r = srstd;
    for (int c = tid; c < C; c += 256)
        out[(size_t)row * C + c] = (X[(size_t)row * C + c] - m) * r * g[c] + bb[c];
}

// =====================================================================
// Host orchestration
// =====================================================================
extern "C" void kernel_launch(void* const* d_in, const int* in_sizes, int n_in,
                              void* d_out, int out_size, void* d_ws, size_t ws_size,
                              hipStream_t stream)
{
    const float* features = (const float*)d_in[0];
    const int*   mask     = (const int*)  d_in[1];
    const float* latents  = (const float*)d_in[2];
    const float* W_down   = (const float*)d_in[3];
    const float* nin_g    = (const float*)d_in[4];
    const float* nin_b    = (const float*)d_in[5];
    const float* lf_g     = (const float*)d_in[6];
    const float* lf_b     = (const float*)d_in[7];
    const float* ll_g     = (const float*)d_in[8];
    const float* ll_b     = (const float*)d_in[9];
    const float* Wq       = (const float*)d_in[10];
    const float* Wk       = (const float*)d_in[11];
    const float* Wv       = (const float*)d_in[12];
    const float* ffln_g   = (const float*)d_in[13];
    const float* ffln_b   = (const float*)d_in[14];
    const float* ff_w1    = (const float*)d_in[15];
    const float* ff_w2    = (const float*)d_in[16];
    const float* nout_g   = (const float*)d_in[17];
    const float* nout_b   = (const float*)d_in[18];
    float* out = (float*)d_out;

    const int BF = B_ * F_;   // 65536
    const int BQ = B_ * Q_;   // 2048

    char* wsb = (char*)d_ws;
    size_t off = 0;
    #define CARVE(name, bytes) void* name = (void*)(wsb + off); \
            off += (((size_t)(bytes)) + 255) & ~(size_t)255;

    CARVE(p_stats_feat, (size_t)BF * 2 * 4);
    CARVE(p_fbuf,       (size_t)BF * INNER_ * 2);
    CARVE(p_stats_f,    (size_t)BF * 2 * 4);
    CARVE(p_Kb,         (size_t)BF * INNER_ * 2);
    CARVE(p_Vb,         (size_t)BF * INNER_ * 2);
    CARVE(p_xbuf,       (size_t)BQ * INNER_ * 4);
    CARVE(p_stats_x,    (size_t)BQ * 2 * 4);
    CARVE(p_qb,         (size_t)BQ * INNER_ * 2);
    CARVE(p_attno,      (size_t)BQ * INNER_ * 4);
    CARVE(p_stats_a,    (size_t)BQ * 2 * 4);
    CARVE(p_h1,         (size_t)BQ * MULT_ * INNER_ * 2);
    CARVE(p_Wd,         (size_t)DIM_ * INNER_ * 2);
    CARVE(p_bd,         (size_t)INNER_ * 4);
    CARVE(p_Wqb,        (size_t)INNER_ * INNER_ * 2);
    CARVE(p_bq,         (size_t)INNER_ * 4);
    CARVE(p_Wkb,        (size_t)INNER_ * INNER_ * 2);
    CARVE(p_bk,         (size_t)INNER_ * 4);
    CARVE(p_Wvb,        (size_t)INNER_ * INNER_ * 2);
    CARVE(p_bv,         (size_t)INNER_ * 4);
    CARVE(p_W1b,        (size_t)INNER_ * MULT_ * INNER_ * 2);
    CARVE(p_b1,         (size_t)MULT_ * INNER_ * 4);
    CARVE(p_W2b,        (size_t)MULT_ * INNER_ * INNER_ * 2);
    #undef CARVE

    float* stats_feat = (float*)p_stats_feat;
    u16*   fbuf       = (u16*)p_fbuf;
    float* stats_f    = (float*)p_stats_f;
    u16*   Kb         = (u16*)p_Kb;
    u16*   Vb         = (u16*)p_Vb;
    float* xbuf       = (float*)p_xbuf;
    float* stats_x    = (float*)p_stats_x;
    u16*   qb         = (u16*)p_qb;
    float* attno      = (float*)p_attno;
    float* stats_a    = (float*)p_stats_a;
    u16*   h1         = (u16*)p_h1;
    u16*   Wd         = (u16*)p_Wd;   float* bd = (float*)p_bd;
    u16*   Wqb        = (u16*)p_Wqb;  float* bq = (float*)p_bq;
    u16*   Wkb        = (u16*)p_Wkb;  float* bk = (float*)p_bk;
    u16*   Wvb        = (u16*)p_Wvb;  float* bv = (float*)p_bv;
    u16*   W1b        = (u16*)p_W1b;  float* b1 = (float*)p_b1;
    u16*   W2b        = (u16*)p_W2b;

    const float qscale = 0.125f;  // DH^-0.5

    // ---- f = LN(features; nin) @ W_down   (LN folded into weights) ----
    rowstats_kernel<false><<<BF, 256, 0, stream>>>(features, stats_feat, DIM_);
    scale_w_kernel<<<(DIM_ * INNER_ + 255) / 256, 256, 0, stream>>>(W_down, nin_g, Wd, DIM_, INNER_, 1.f);
    bias_gemv_kernel<<<INNER_ / 256, 256, 0, stream>>>(W_down, nin_b, bd, DIM_, INNER_, 1.f);
    gemm_wmma_kernel<false, true, true, true, false, false>
        <<<dim3(BF / 128, INNER_ / 128), 256, 0, stream>>>(
            features, stats_feat, Wd, bd, fbuf, nullptr, BF, INNER_, DIM_);

    // ---- materialize fhat = (f - mean)/std  once (layer-independent) ----
    rowstats_kernel<true><<<BF, 256, 0, stream>>>(fbuf, stats_f, INNER_);
    norm_bf16_kernel<<<(BF * INNER_) / 256, 256, 0, stream>>>(fbuf, stats_f);

    // ---- x = broadcast(latents) ----
    bcast_kernel<<<(BQ * INNER_ + 255) / 256, 256, 0, stream>>>(latents, xbuf, BQ * INNER_, Q_ * INNER_);

    for (int i = 0; i < DEPTH_; i++) {
        const float* Wqi = Wq + (size_t)i * INNER_ * INNER_;
        const float* Wki = Wk + (size_t)i * INNER_ * INNER_;
        const float* Wvi = Wv + (size_t)i * INNER_ * INNER_;
        const float* W1i = ff_w1 + (size_t)i * INNER_ * MULT_ * INNER_;
        const float* W2i = ff_w2 + (size_t)i * MULT_ * INNER_ * INNER_;
        const float* lfg = lf_g + (size_t)i * INNER_;
        const float* lfb = lf_b + (size_t)i * INNER_;
        const float* llg = ll_g + (size_t)i * INNER_;
        const float* llb = ll_b + (size_t)i * INNER_;
        const float* ffg = ffln_g + (size_t)i * INNER_;
        const float* ffb = ffln_b + (size_t)i * INNER_;

        // weight folding (LN gains -> weights; LN shifts -> bias; q scale folded)
        scale_w_kernel<<<(INNER_ * INNER_ + 255) / 256, 256, 0, stream>>>(Wqi, llg, Wqb, INNER_, INNER_, qscale);
        bias_gemv_kernel<<<INNER_ / 256, 256, 0, stream>>>(Wqi, llb, bq, INNER_, INNER_, qscale);
        scale_w_kernel<<<(INNER_ * INNER_ + 255) / 256, 256, 0, stream>>>(Wki, lfg, Wkb, INNER_, INNER_, 1.f);
        bias_gemv_kernel<<<INNER_ / 256, 256, 0, stream>>>(Wki, lfb, bk, INNER_, INNER_, 1.f);
        scale_w_kernel<<<(INNER_ * INNER_ + 255) / 256, 256, 0, stream>>>(Wvi, lfg, Wvb, INNER_, INNER_, 1.f);
        bias_gemv_kernel<<<INNER_ / 256, 256, 0, stream>>>(Wvi, lfb, bv, INNER_, INNER_, 1.f);
        scale_w_kernel<<<(INNER_ * MULT_ * INNER_ + 255) / 256, 256, 0, stream>>>(W1i, ffg, W1b, INNER_, MULT_ * INNER_, 1.f);
        bias_gemv_kernel<<<(MULT_ * INNER_) / 256, 256, 0, stream>>>(W1i, ffb, b1, INNER_, MULT_ * INNER_, 1.f);
        scale_w_kernel<<<(MULT_ * INNER_ * INNER_ + 255) / 256, 256, 0, stream>>>(W2i, nullptr, W2b, MULT_ * INNER_, INNER_, 1.f);

        // q = LN(x; ll) @ Wq * scale  (stats + folded weights)
        rowstats_kernel<false><<<BQ, 256, 0, stream>>>(xbuf, stats_x, INNER_);
        gemm_wmma_kernel<false, true, true, true, false, false>
            <<<dim3(BQ / 128, INNER_ / 128), 256, 0, stream>>>(
                xbuf, stats_x, Wqb, bq, qb, nullptr, BQ, INNER_, INNER_);

        // k, v = fhat @ {Wk', Wv'} + bias   (TDM-staged, double-buffered)
        gemm_wmma_kernel<true, false, true, true, false, false>
            <<<dim3(BF / 128, INNER_ / 128), 256, 0, stream>>>(
                fbuf, nullptr, Wkb, bk, Kb, nullptr, BF, INNER_, INNER_);
        gemm_wmma_kernel<true, false, true, true, false, false>
            <<<dim3(BF / 128, INNER_ / 128), 256, 0, stream>>>(
                fbuf, nullptr, Wvb, bv, Vb, nullptr, BF, INNER_, INNER_);

        // masked attention -> (B, Q, INNER) in bqhd layout
        attn_kernel<<<dim3(H_, B_), 128, 0, stream>>>(qb, Kb, Vb, mask, attno);

        // h = relu(LN(out; ffln) @ ff_w1) @ ff_w2 ; x += h
        rowstats_kernel<false><<<BQ, 256, 0, stream>>>(attno, stats_a, INNER_);
        gemm_wmma_kernel<false, true, true, true, true, false>
            <<<dim3(BQ / 128, (MULT_ * INNER_) / 128), 256, 0, stream>>>(
                attno, stats_a, W1b, b1, h1, nullptr, BQ, MULT_ * INNER_, INNER_);
        gemm_wmma_kernel<true, false, false, false, false, true>
            <<<dim3(BQ / 128, INNER_ / 128), 256, 0, stream>>>(
                h1, nullptr, W2b, nullptr, xbuf, xbuf, BQ, INNER_, MULT_ * INNER_);
    }

    // ---- final LN ----
    ln_final_kernel<<<BQ, 256, 0, stream>>>(xbuf, nout_g, nout_b, out, INNER_);
    (void)in_sizes; (void)n_in; (void)out_size; (void)ws_size;
}
